// GraphAttentionNetworkInductive_34050500722965
// MI455X (gfx1250) — matrix operand order repaired and checked by reading, not compiled, not run on hardware
//
#include <hip/hip_runtime.h>
#include <math.h>

// ---------------------------------------------------------------------------
// GAT (3 layers) for MI455X / gfx1250, wave32, WMMA f16->f32 GEMMs with
// bank-conflict-free LDS-staged B panels.
// N=30000, E=480000, F=256, dims 512/512/726 (6x121).
// ---------------------------------------------------------------------------

#define NN 30000
#define EE 480000
#define FF 256
#define OUTD 121
#define NEG_SLOPE 0.2f
#define LDS_PAD 4                     // dword skew per staged column

typedef _Float16 half_t;
typedef __attribute__((ext_vector_type(16))) _Float16 v16h;
typedef __attribute__((ext_vector_type(8)))  float    v8f;

union FragU { unsigned u[8]; uint4 q[2]; v16h h; };

// ---------------------------------------------------------------------------
// f32 -> f16 convert
// ---------------------------------------------------------------------------
__global__ void cvt_f32_f16(const float* __restrict__ x, half_t* __restrict__ y, int n) {
  int t = blockIdx.x * blockDim.x + threadIdx.x;
  if (t < n) y[t] = (half_t)x[t];
}

// ---------------------------------------------------------------------------
// Build B-operand buffer: Bt[c][k] = W[h][k][u] with c = h*U+u  (f16),
// zero padded up to Cpad columns. W is [H, K, U] row-major f32.
// ---------------------------------------------------------------------------
__global__ void prep_bt(const float* __restrict__ W, half_t* __restrict__ Bt,
                        int H, int K, int U, int Cpad) {
  int t = blockIdx.x * blockDim.x + threadIdx.x;
  if (t >= Cpad * K) return;
  int c = t / K, k = t % K;
  float v = 0.0f;
  if (c < H * U) {
    int h = c / U, u = c % U;
    v = W[((size_t)h * K + k) * U + u];
  }
  Bt[(size_t)c * K + k] = (half_t)v;
}

// ---------------------------------------------------------------------------
// WMMA GEMM:  C[M,Cols] = A[M,K](f16) * B, B given as Bt[Cpad,K](f16)
// (Bt row c = column c of B).
// Grid: x = ceil(tilesM/8) (one 16-row tile per wave, 8 waves/block),
//       y = Cpad/64 (one 64-column strip per block, staged in LDS with a
//       4-dword skew per column so b128 LDS reads are bank-conflict-free).
// K multiple of 32 (<=512), M multiple of 16, Cpad multiple of 64.
// Fragment layouts (contiguous k order, not the ISA table order): A lane->M,
// B lane->N; element s of lane half h holds k = k0 + 16h + s in both A and B.
// ---------------------------------------------------------------------------
__global__ void wmma_gemm(const half_t* __restrict__ A, const half_t* __restrict__ Bt,
                          float* __restrict__ C, int M, int K, int Cols) {
  __shared__ unsigned sB[64 * (256 + LDS_PAD)];   // 64 cols x (K/2+4) dwords, 65KB max
  int kd   = K >> 1;                              // dwords per column of Bt
  int padd = kd + LDS_PAD;                        // skewed LDS column stride (dwords)

  // cooperative stage: 64 columns of this block's strip -> LDS (uint4 chunks)
  int n0 = blockIdx.y << 6;
  const uint4* gB = (const uint4*)(Bt + (size_t)n0 * K);
  int kq = kd >> 2;                               // uint4 chunks per column
  for (int idx = threadIdx.x; idx < 64 * kq; idx += blockDim.x) {
    int c = idx / kq, i = idx % kq;
    ((uint4*)(sB + (size_t)c * padd))[i] = gB[idx];
  }
  __syncthreads();

  int wave = threadIdx.x >> 5;
  int lane = threadIdx.x & 31;
  int tm = blockIdx.x * 8 + wave;
  if (tm >= (M >> 4)) return;                     // wave-uniform: EXEC all-1s below
  int m0 = tm << 4;
  int l16 = lane & 15, half = lane >> 4;

  v8f acc0 = {0,0,0,0,0,0,0,0};
  v8f acc1 = acc0, acc2 = acc0, acc3 = acc0;

  const unsigned* Arow = (const unsigned*)(A + (size_t)(m0 + l16) * K);
  const uint4* Bq0 = (const uint4*)(sB + (size_t)(l16 +  0) * padd);
  const uint4* Bq1 = (const uint4*)(sB + (size_t)(l16 + 16) * padd);
  const uint4* Bq2 = (const uint4*)(sB + (size_t)(l16 + 32) * padd);
  const uint4* Bq3 = (const uint4*)(sB + (size_t)(l16 + 48) * padd);

  for (int k0 = 0; k0 < K; k0 += 32) {
    // A fragment: k = k0 + half*16 + 2*v
    FragU a;
    int ab = (k0 >> 1) + half * 8;
#pragma unroll
    for (int v = 0; v < 4; ++v) {
      a.u[v]     = Arow[ab + v];
      a.u[v + 4] = Arow[ab + 4 + v];
    }
    // B fragments from LDS: k = k0 + half*16 + 2v -> dwords bb..bb+7, 2x b128
    int bq = (k0 >> 3) + half * 2;                // uint4 index = (k0/2 + half*8)/4
    FragU b0, b1, b2, b3;
    b0.q[0] = Bq0[bq]; b0.q[1] = Bq0[bq + 1];
    b1.q[0] = Bq1[bq]; b1.q[1] = Bq1[bq + 1];
    b2.q[0] = Bq2[bq]; b2.q[1] = Bq2[bq + 1];
    b3.q[0] = Bq3[bq]; b3.q[1] = Bq3[bq + 1];
    acc0 = __builtin_amdgcn_wmma_f32_16x16x32_f16(false, a.h, false, b0.h, (short)0, acc0, false, false);
    acc1 = __builtin_amdgcn_wmma_f32_16x16x32_f16(false, a.h, false, b1.h, (short)0, acc1, false, false);
    acc2 = __builtin_amdgcn_wmma_f32_16x16x32_f16(false, a.h, false, b2.h, (short)0, acc2, false, false);
    acc3 = __builtin_amdgcn_wmma_f32_16x16x32_f16(false, a.h, false, b3.h, (short)0, acc3, false, false);
  }

  // C/D layout: VGPR d, lane -> row m0 + d + half*8, col n0 + l16 (+16*s)
  v8f accs[4] = {acc0, acc1, acc2, acc3};
#pragma unroll
  for (int s = 0; s < 4; ++s) {
    int c = n0 + l16 + 16 * s;
    if (c < Cols) {
#pragma unroll
      for (int d = 0; d < 8; ++d)
        C[(size_t)(m0 + d + half * 8) * Cols + c] = accs[s][d];
    }
  }
}

// ---------------------------------------------------------------------------
// es[n,h] = <h[n,h,:], a_s[h,:]> ; ed likewise
// ---------------------------------------------------------------------------
__global__ void attn_scores(const float* __restrict__ hl, const float* __restrict__ as_,
                            const float* __restrict__ ad_, float* __restrict__ es,
                            float* __restrict__ ed, int Nn, int H, int U) {
  int t = blockIdx.x * blockDim.x + threadIdx.x;
  if (t >= Nn * H) return;
  int n = t / H, h = t % H;
  const float* hv = hl + (size_t)n * H * U + (size_t)h * U;
  const float* a1 = as_ + (size_t)h * U;
  const float* a2 = ad_ + (size_t)h * U;
  float s = 0.0f, d = 0.0f;
  for (int u = 0; u < U; ++u) { s += hv[u] * a1[u]; d += hv[u] * a2[u]; }
  es[t] = s; ed[t] = d;
}

__global__ void init_u32(unsigned* __restrict__ p, int n, unsigned v) {
  int t = blockIdx.x * blockDim.x + threadIdx.x;
  if (t < n) p[t] = v;
}

__device__ __forceinline__ float lrelu(float x) { return x >= 0.0f ? x : NEG_SLOPE * x; }

// orderable-u32 mapping for float atomicMax
__device__ __forceinline__ unsigned f2ord(float f) {
  unsigned u = __float_as_uint(f);
  return (u & 0x80000000u) ? ~u : (u | 0x80000000u);
}

// segment max of edge logits into m[dst,h] (orderable-u32)
__global__ void edge_max(const int* __restrict__ src, const int* __restrict__ dst,
                         const float* __restrict__ es, const float* __restrict__ ed,
                         unsigned* __restrict__ m, int E, int H) {
  int t = blockIdx.x * blockDim.x + threadIdx.x;
  if (t >= E * H) return;
  int e = t / H, h = t % H;
  float logit = lrelu(es[src[e] * H + h] + ed[dst[e] * H + h]);
  atomicMax(&m[dst[e] * H + h], f2ord(logit));
}

// decode orderable-u32 back to float in place; non-finite -> 0 (ref semantics)
__global__ void m_fixup(unsigned* __restrict__ m, int n) {
  int t = blockIdx.x * blockDim.x + threadIdx.x;
  if (t >= n) return;
  unsigned u = m[t];
  float f = (u & 0x80000000u) ? __uint_as_float(u & 0x7fffffffu) : __uint_as_float(~u);
  if (!isfinite(f)) f = 0.0f;
  ((float*)m)[t] = f;
}

// one wave per (edge, head): ex = exp(logit - m[dst]); denom += ex;
// agg[dst,h,:] += ex * h[src,h,:]
__global__ void edge_msg(const int* __restrict__ src, const int* __restrict__ dst,
                         const float* __restrict__ es, const float* __restrict__ ed,
                         const float* __restrict__ mf, const float* __restrict__ hl,
                         float* __restrict__ denom, float* __restrict__ agg,
                         int E, int H, int U) {
  int gid = blockIdx.x * blockDim.x + threadIdx.x;
  int wave = gid >> 5, lane = gid & 31;
  if (wave >= E * H) return;
  int e = wave / H, h = wave % H;
  int s = src[e], d = dst[e];
  float logit = lrelu(es[s * H + h] + ed[d * H + h]);
  float ex = expf(logit - mf[d * H + h]);
  if (lane == 0) atomicAdd(&denom[d * H + h], ex);
  const float* hs = hl + (size_t)s * H * U + (size_t)h * U;
  float* ag = agg + (size_t)d * H * U + (size_t)h * U;
  for (int u = lane; u < U; u += 32)
    atomicAdd(&ag[u], ex * hs[u]);
}

// concat merge + residual + ELU; writes f32 and f16 copies
__global__ void finalize_concat(const float* __restrict__ agg, const float* __restrict__ denom,
                                const float* __restrict__ res, float* __restrict__ outf,
                                half_t* __restrict__ outh, int Nn, int H, int U) {
  int t = blockIdx.x * blockDim.x + threadIdx.x;
  if (t >= Nn * H * U) return;
  int n = t / (H * U);
  int c = t % (H * U);
  int h = c / U;
  float v = agg[t] / fmaxf(denom[n * H + h], 1e-9f) + res[t];
  v = v > 0.0f ? v : (expf(v) - 1.0f);              // ELU
  outf[t] = v;
  outh[t] = (half_t)v;
}

// average merge + residual (logits, no activation)
__global__ void finalize_avg(const float* __restrict__ agg, const float* __restrict__ denom,
                             const float* __restrict__ r3, float* __restrict__ out,
                             int Nn, int H, int U) {
  int t = blockIdx.x * blockDim.x + threadIdx.x;
  if (t >= Nn * U) return;
  int n = t / U, u = t % U;
  float s = 0.0f;
  for (int h = 0; h < H; ++h)
    s += agg[((size_t)n * H + h) * U + u] / fmaxf(denom[n * H + h], 1e-9f);
  out[t] = s / (float)H + r3[t];
}

// ---------------------------------------------------------------------------
// host side
// ---------------------------------------------------------------------------
static inline int blks(long long n, int b) { return (int)((n + b - 1) / b); }

extern "C" void kernel_launch(void* const* d_in, const int* in_sizes, int n_in,
                              void* d_out, int out_size, void* d_ws, size_t ws_size,
                              hipStream_t stream) {
  (void)in_sizes; (void)n_in; (void)out_size; (void)ws_size;
  const float* x   = (const float*)d_in[0];
  const int*   src = (const int*)d_in[1];
  const int*   dst = (const int*)d_in[2];
  const float* W1  = (const float*)d_in[3];
  const float* a1s = (const float*)d_in[4];
  const float* a1d = (const float*)d_in[5];
  const float* Wr1 = (const float*)d_in[6];
  const float* W2  = (const float*)d_in[7];
  const float* a2s = (const float*)d_in[8];
  const float* a2d = (const float*)d_in[9];
  const float* W3  = (const float*)d_in[10];
  const float* a3s = (const float*)d_in[11];
  const float* a3d = (const float*)d_in[12];
  const float* Wr3 = (const float*)d_in[13];
  float* out = (float*)d_out;

  // scratch carve-out
  char* ws = (char*)d_ws;
  size_t off = 0;
  auto alloc = [&](size_t bytes) -> void* {
    off = (off + 255) & ~(size_t)255;
    void* p = ws + off;
    off += bytes;
    return p;
  };
  half_t* xh   = (half_t*)alloc((size_t)NN * FF * 2);       // x in f16
  half_t* h1h  = (half_t*)alloc((size_t)NN * 512 * 2);      // h1 in f16
  half_t* h2h  = (half_t*)alloc((size_t)NN * 512 * 2);      // h2 in f16
  float*  h1f  = (float*)alloc((size_t)NN * 512 * 4);       // h1 f32 (reused as h2 f32)
  float*  hl   = (float*)alloc((size_t)NN * 726 * 4);       // per-layer head features
  float*  agg  = (float*)alloc((size_t)NN * 726 * 4);       // message accumulator
  float*  res  = (float*)alloc((size_t)NN * 512 * 4);       // residual projection (r1 / r3)
  float*  es   = (float*)alloc((size_t)NN * 6 * 4);
  float*  ed   = (float*)alloc((size_t)NN * 6 * 4);
  unsigned* m  = (unsigned*)alloc((size_t)NN * 6 * 4);      // seg-max (u32 then f32)
  float*  den  = (float*)alloc((size_t)NN * 6 * 4);
  half_t* Bt   = (half_t*)alloc((size_t)768 * 512 * 2);     // largest weight operand

  const unsigned ORD_NEG_INF = 0x007FFFFFu;                 // f2ord(-inf)

  auto gemm = [&](const half_t* A, const half_t* B, float* C, int M, int K, int Cols, int Cpad) {
    dim3 g((unsigned)((M / 16 + 7) / 8), (unsigned)(Cpad / 64));
    wmma_gemm<<<g, 256, 0, stream>>>(A, B, C, M, K, Cols);
  };
  auto edge_phase = [&](const float* as_, const float* ad_, int H, int U) {
    attn_scores<<<blks((long long)NN * H, 256), 256, 0, stream>>>(hl, as_, ad_, es, ed, NN, H, U);
    init_u32<<<blks((long long)NN * H, 256), 256, 0, stream>>>(m, NN * H, ORD_NEG_INF);
    hipMemsetAsync(den, 0, (size_t)NN * H * 4, stream);
    hipMemsetAsync(agg, 0, (size_t)NN * H * U * 4, stream);
    edge_max<<<blks((long long)EE * H, 256), 256, 0, stream>>>(src, dst, es, ed, m, EE, H);
    m_fixup<<<blks((long long)NN * H, 256), 256, 0, stream>>>(m, NN * H);
    edge_msg<<<blks((long long)EE * H * 32, 256), 256, 0, stream>>>(
        src, dst, es, ed, (const float*)m, hl, den, agg, EE, H, U);
  };

  // ---- prep -------------------------------------------------------------
  cvt_f32_f16<<<blks((long long)NN * FF, 256), 256, 0, stream>>>(x, xh, NN * FF);

  // ---- layer 1: 4 heads x 128, concat, projected residual, ELU ----------
  prep_bt<<<blks(512 * 256, 256), 256, 0, stream>>>(W1, Bt, 4, 256, 128, 512);
  gemm(xh, Bt, hl, NN, 256, 512, 512);
  prep_bt<<<blks(512 * 256, 256), 256, 0, stream>>>(Wr1, Bt, 1, 256, 512, 512);
  gemm(xh, Bt, res, NN, 256, 512, 512);
  edge_phase(a1s, a1d, 4, 128);
  finalize_concat<<<blks((long long)NN * 512, 256), 256, 0, stream>>>(
      agg, den, res, h1f, h1h, NN, 4, 128);

  // ---- layer 2: 4 heads x 128, concat, identity residual, ELU -----------
  prep_bt<<<blks(512 * 512, 256), 256, 0, stream>>>(W2, Bt, 4, 512, 128, 512);
  gemm(h1h, Bt, hl, NN, 512, 512, 512);
  edge_phase(a2s, a2d, 4, 128);
  finalize_concat<<<blks((long long)NN * 512, 256), 256, 0, stream>>>(
      agg, den, h1f /*identity res*/, h1f /*in-place f32*/, h2h, NN, 4, 128);

  // ---- layer 3: 6 heads x 121, avg, projected residual, logits ----------
  prep_bt<<<blks(768 * 512, 256), 256, 0, stream>>>(W3, Bt, 6, 512, 121, 768);
  gemm(h2h, Bt, hl, NN, 512, 726, 768);
  prep_bt<<<blks(128 * 512, 256), 256, 0, stream>>>(Wr3, Bt, 1, 512, 121, 128);
  gemm(h2h, Bt, res, NN, 512, 121, 128);
  edge_phase(a3s, a3d, 6, 121);
  finalize_avg<<<blks((long long)NN * OUTD, 256), 256, 0, stream>>>(
      agg, den, res, out, NN, 6, OUTD);
}
